// self_atten_15848429322428
// MI455X (gfx1250) — matrix operand order repaired and checked by reading, not compiled, not run on hardware
//
#include <hip/hip_runtime.h>

// ---------------------------------------------------------------------------
// Problem constants (fixed by the reference)
// ---------------------------------------------------------------------------
typedef unsigned short ushort_t;
typedef __attribute__((ext_vector_type(16))) __bf16 v16bf;
typedef __attribute__((ext_vector_type(8)))  float  v8f;
typedef __attribute__((ext_vector_type(4)))  unsigned int v4u;
typedef __attribute__((ext_vector_type(4)))  int i32x4;
typedef __attribute__((ext_vector_type(8)))  int i32x8;

constexpr int Bb = 2;
constexpr int Ss = 2048;
constexpr int Dm = 1024;
constexpr int Hh = 16;
constexpr int HD = 64;           // head dim
constexpr int Mrows = Bb * Ss;   // 4096
constexpr float NEG = -1.0e9f;

#if defined(__has_builtin)
#if __has_builtin(__builtin_amdgcn_tensor_load_to_lds)
#define USE_TDM 1
#else
#define USE_TDM 0
#endif
#else
#define USE_TDM 0
#endif

// ---------------------------------------------------------------------------
// bf16 helpers (round-to-nearest-even on raw u16 storage)
// ---------------------------------------------------------------------------
__device__ __forceinline__ ushort_t f2bf(float f) {
  unsigned int u = __float_as_uint(f);
  unsigned int r = u + 0x7FFFu + ((u >> 16) & 1u);
  return (ushort_t)(r >> 16);
}

union Frag16 {
  struct { v4u lo, hi; } p;   // 2 x 16B = 32B
  v16bf v;                    // 16 bf16
};

// A-operand fragment, 16x32 (MxK) bf16, per CDNA5 ISA layout:
//   lane r=lane&15 holds row M=r; half g=(lane>>4):
//   elems 0..7 = K(g*8..g*8+7), elems 8..15 = K(16+g*8..16+g*8+7)
__device__ __forceinline__ v16bf load_a_frag(const ushort_t* A, int lda,
                                             int row, int k0, int lane) {
  int r = lane & 15;
  int g = (lane >> 4) * 8;
  const ushort_t* p = A + (size_t)(row + r) * lda + (k0 + g);
  Frag16 f;
  f.p.lo = *(const v4u*)(p);
  f.p.hi = *(const v4u*)(p + 16);
  return f.v;
}

// B-operand fragment, 32x16 (KxN) bf16, B[k][n] taken from Bt stored [N,K]:
//   lane c=lane&15 -> column N=c; half g=(lane>>4)*8 -> K g..g+7, 16+g..16+g+7.
__device__ __forceinline__ v16bf load_b_frag(const ushort_t* Bt, int ldb,
                                             int k0, int col, int lane) {
  int c = lane & 15;
  int g = (lane >> 4) * 8;
  const ushort_t* p = Bt + (size_t)(col + c) * ldb + (k0 + g);
  Frag16 f;
  f.p.lo = *(const v4u*)(p);
  f.p.hi = *(const v4u*)(p + 16);
  return f.v;
}

__device__ __forceinline__ v8f wmma_bf16(v16bf a, v16bf b, v8f c) {
  return __builtin_amdgcn_wmma_f32_16x16x32_bf16(
      false, a, false, b, (short)0, c, false, false);
}

// ---------------------------------------------------------------------------
// Tensor Data Mover: 2D tile (tile_d0 x tile_d1, 2-byte elems) -> LDS.
// D# packed per CDNA5 ISA (groups 0/1).
// Semantics are best-effort per the ISA bit layout; compile-only environment.
// ---------------------------------------------------------------------------
#if USE_TDM
__device__ __forceinline__ void tdm_load_2d(unsigned lds_off, const void* gaddr,
                                            unsigned tensor_d0, unsigned tensor_d1,
                                            unsigned tile_d0, unsigned tile_d1,
                                            unsigned long long stride0) {
  unsigned long long ga = (unsigned long long)(size_t)gaddr;
  v4u g0;
  g0[0] = 1u;                                            // count=1 (valid user D#)
  g0[1] = lds_off;                                       // lds_addr (bytes)
  g0[2] = (unsigned)(ga & 0xFFFFFFFFu);                  // global_addr[31:0]
  g0[3] = (unsigned)((ga >> 32) & 0x01FFFFFFu) | (2u << 30);  // addr[56:32] | type=2
  i32x8 g1;
  g1[0] = (int)(1u << 16);                               // data_size=1 (2 bytes)
  g1[1] = (int)((tensor_d0 & 0xFFFFu) << 16);            // tensor_dim0[15:0]
  g1[2] = (int)(((tensor_d0 >> 16) & 0xFFFFu) |
                ((tensor_d1 & 0xFFFFu) << 16));          // dim0[31:16] | dim1[15:0]
  g1[3] = (int)(((tensor_d1 >> 16) & 0xFFFFu) |
                ((tile_d0 & 0xFFFFu) << 16));            // dim1[31:16] | tile_dim0
  g1[4] = (int)(tile_d1 & 0xFFFFu);                      // tile_dim1 (tile_dim2=0)
  g1[5] = (int)(unsigned)(stride0 & 0xFFFFFFFFull);      // dim0_stride[31:0]
  g1[6] = (int)(unsigned)((stride0 >> 32) & 0xFFFFull);  // dim0_stride[47:32]
  g1[7] = 0;
  i32x4 z4 = {0, 0, 0, 0};
#if __clang_major__ >= 23
  i32x8 z8 = {0, 0, 0, 0, 0, 0, 0, 0};
  __builtin_amdgcn_tensor_load_to_lds(g0, g1, z4, z4, z8, 0);
#else
  __builtin_amdgcn_tensor_load_to_lds(g0, g1, z4, z4, 0);
#endif
}
#endif

// Fallback cooperative staging: 256 threads copy a 128x32 bf16 panel.
__device__ __forceinline__ void stage_panel_fallback(ushort_t* dst,
                                                     const ushort_t* src,
                                                     int ld, int tid) {
  int r = tid >> 1, half = tid & 1;
  const v4u* s = (const v4u*)(src + (size_t)r * ld + half * 16);
  v4u* d = (v4u*)(dst + r * 32 + half * 16);
  d[0] = s[0];
  d[1] = s[1];
}

// ---------------------------------------------------------------------------
// fp32 -> bf16 conversion (x)
// ---------------------------------------------------------------------------
__global__ __launch_bounds__(256) void convert_f32_bf16(
    const float* __restrict__ in, ushort_t* __restrict__ out, int n) {
  int i = blockIdx.x * blockDim.x + threadIdx.x;
  int stride = gridDim.x * blockDim.x;
  for (; i < n; i += stride) out[i] = f2bf(in[i]);
}

// W[K][N] fp32 -> Wt[N][K] bf16 (LDS tile transpose)
__global__ __launch_bounds__(256) void transpose_convert(
    const float* __restrict__ W, ushort_t* __restrict__ Wt) {
  __shared__ float tile[32][33];
  int nb = blockIdx.x * 32, kb = blockIdx.y * 32;
  int tx = threadIdx.x & 31, ty = threadIdx.x >> 5;
#pragma unroll
  for (int i = ty; i < 32; i += 8)
    tile[i][tx] = W[(size_t)(kb + i) * Dm + (nb + tx)];
  __syncthreads();
#pragma unroll
  for (int i = ty; i < 32; i += 8)
    Wt[(size_t)(nb + i) * Dm + (kb + tx)] = f2bf(tile[tx][i]);
}

// ---------------------------------------------------------------------------
// WMMA GEMM with TDM-staged, double-buffered LDS K-panels.
//   out = relu(A[M,K]bf16 @ Wt[N,K]bf16^T + bias)
//   MODE 0: bf16 [M,N]; MODE 1: bf16 vt[B][H][HD][S]; MODE 2: fp32 [M,N]
// Block = 256 thr (8 waves), tile 128x128; wave tile 64x32 = 4x2 WMMAs.
// ---------------------------------------------------------------------------
template <int MODE>
__global__ __launch_bounds__(256) void gemm_bias_relu(
    const ushort_t* __restrict__ A, const ushort_t* __restrict__ Bt,
    const float* __restrict__ bias, void* __restrict__ out,
    int M, int N, int K) {
  __shared__ ushort_t sA[2][128 * 32];   // 8 KB per buffer
  __shared__ ushort_t sB[2][128 * 32];

  int tid = threadIdx.x;
  int lane = tid & 31;
  int w = tid >> 5;
  int wr = w >> 2, wc = w & 3;           // wave grid 2x4 over 128x128 tile
  int blk_row = blockIdx.y * 128;
  int blk_col = blockIdx.x * 128;

  v8f acc[4][2] = {};
  int nsteps = K / 32;

#if USE_TDM
  // Prologue: DMA first K-panels into buffer 0.
  if (w == 0) {
    tdm_load_2d((unsigned)(size_t)&sA[0][0], A + (size_t)blk_row * K,
                (unsigned)K, (unsigned)M, 32u, 128u, (unsigned long long)K);
    tdm_load_2d((unsigned)(size_t)&sB[0][0], Bt + (size_t)blk_col * K,
                (unsigned)K, (unsigned)N, 32u, 128u, (unsigned long long)K);
    __builtin_amdgcn_s_wait_tensorcnt(0);
  }
  __syncthreads();
#else
  stage_panel_fallback(&sA[0][0], A + (size_t)blk_row * K, K, tid);
  stage_panel_fallback(&sB[0][0], Bt + (size_t)blk_col * K, K, tid);
  __syncthreads();
#endif

  for (int s = 0; s < nsteps; ++s) {
    int cur = s & 1;
    int k0 = s * 32;

#if USE_TDM
    // Kick off the DMA for the next K-panel into the other buffer while we
    // compute on this one (TENSORcnt overlap).
    if (s + 1 < nsteps && w == 0) {
      int nxt = cur ^ 1;
      tdm_load_2d((unsigned)(size_t)&sA[nxt][0],
                  A + (size_t)blk_row * K + (k0 + 32),
                  (unsigned)K, (unsigned)M, 32u, 128u, (unsigned long long)K);
      tdm_load_2d((unsigned)(size_t)&sB[nxt][0],
                  Bt + (size_t)blk_col * K + (k0 + 32),
                  (unsigned)K, (unsigned)N, 32u, 128u, (unsigned long long)K);
    }
#endif

    // Fragments from LDS (ds_load_b128), per documented A/B layouts.
    int r = lane & 15;
    int g8 = (lane >> 4) * 8, g16 = (lane >> 4) * 16;
    v16bf a[4], b[2];
#pragma unroll
    for (int mi = 0; mi < 4; ++mi) {
      const ushort_t* p = &sA[cur][(wr * 64 + mi * 16 + r) * 32 + g8];
      Frag16 f;
      f.p.lo = *(const v4u*)(p);
      f.p.hi = *(const v4u*)(p + 16);
      a[mi] = f.v;
    }
#pragma unroll
    for (int ni = 0; ni < 2; ++ni) {
      const ushort_t* p = &sB[cur][(wc * 32 + ni * 16 + r) * 32 + g8];
      Frag16 f;
      f.p.lo = *(const v4u*)(p);
      f.p.hi = *(const v4u*)(p + 16);
      b[ni] = f.v;
    }
#pragma unroll
    for (int mi = 0; mi < 4; ++mi)
#pragma unroll
      for (int ni = 0; ni < 2; ++ni)
        acc[mi][ni] = wmma_bf16(a[mi], b[ni], acc[mi][ni]);

#if USE_TDM
    if (w == 0) __builtin_amdgcn_s_wait_tensorcnt(0);
    __syncthreads();
#else
    __syncthreads();
    if (s + 1 < nsteps) {
      stage_panel_fallback(&sA[(s + 1) & 1][0],
                           A + (size_t)blk_row * K + (k0 + 32), K, tid);
      stage_panel_fallback(&sB[(s + 1) & 1][0],
                           Bt + (size_t)blk_col * K + (k0 + 32), K, tid);
      __syncthreads();
    }
#endif
  }

  // Epilogue: C layout -> VGPR j holds M=j+8*(lane>>4), N=lane&15
  int hi = lane >> 4, nl = lane & 15;
#pragma unroll
  for (int mi = 0; mi < 4; ++mi) {
#pragma unroll
    for (int ni = 0; ni < 2; ++ni) {
      int col = blk_col + wc * 32 + ni * 16 + nl;
      float bsv = bias[col];
#pragma unroll
      for (int j = 0; j < 8; ++j) {
        int row = blk_row + wr * 64 + mi * 16 + j + 8 * hi;
        float v = acc[mi][ni][j] + bsv;
        v = fmaxf(v, 0.0f);
        if (MODE == 0) {
          ((ushort_t*)out)[(size_t)row * N + col] = f2bf(v);
        } else if (MODE == 1) {
          int bi = row >> 11;        // row / S
          int s2 = row & (Ss - 1);
          int h  = col >> 6;         // col / HD
          int dd = col & (HD - 1);
          ((ushort_t*)out)[(((size_t)(bi * Hh + h)) * HD + dd) * Ss + s2] = f2bf(v);
        } else {
          ((float*)out)[(size_t)row * N + col] = v;
        }
      }
    }
  }
}

// ---------------------------------------------------------------------------
// Flash attention: one block per (b, h, 64 query rows); 4 waves x 16 rows.
// ---------------------------------------------------------------------------
__global__ __launch_bounds__(128) void attention_kernel(
    const ushort_t* __restrict__ qb, const ushort_t* __restrict__ kb,
    const ushort_t* __restrict__ vt, const int* __restrict__ mask,
    ushort_t* __restrict__ ctx) {
  int qblk = blockIdx.x, h = blockIdx.y, b = blockIdx.z;
  int lane = threadIdx.x & 31;
  int w = threadIdx.x >> 5;
  int q0 = qblk * 64 + w * 16;

  const ushort_t* qhp = qb + (size_t)b * Ss * Dm + h * HD;     // ld = Dm
  const ushort_t* khp = kb + (size_t)b * Ss * Dm + h * HD;     // ld = Dm
  const ushort_t* vtp = vt + ((size_t)(b * Hh + h)) * HD * Ss; // [HD][S]
  const int* mrow = mask + (size_t)b * Ss * Ss;

  __shared__ ushort_t p_lds[4][16 * 64];  // per-wave P staging (8 KB)

  v16bf aq0 = load_a_frag(qhp, Dm, q0, 0, lane);
  v16bf aq1 = load_a_frag(qhp, Dm, q0, 32, lane);

  v8f o[4] = {};
  float m[8], l[8];
#pragma unroll
  for (int j = 0; j < 8; ++j) { m[j] = -3.0e38f; l[j] = 0.0f; }

  const float scale = 0.125f;
  int hi = lane >> 4, nl = lane & 15;

  for (int kc = 0; kc < Ss; kc += 64) {
    if (kc + 64 < Ss) {  // gfx1250 global_prefetch for next chunk
      __builtin_prefetch(khp + (size_t)(kc + 64 + (lane & 15)) * Dm, 0, 3);
      __builtin_prefetch(vtp + (size_t)(lane & 15) * Ss + kc + 64, 0, 3);
    }

    // scores 16x64 = Q(16x64) @ K^T via chained WMMA
    v8f sc[4];
#pragma unroll
    for (int ns = 0; ns < 4; ++ns) {
      int key0 = kc + ns * 16;
      v16bf bk0 = load_b_frag(khp, Dm, 0, key0, lane);
      v16bf bk1 = load_b_frag(khp, Dm, 32, key0, lane);
      v8f c = {};
      c = wmma_bf16(aq0, bk0, c);
      c = wmma_bf16(aq1, bk1, c);
      sc[ns] = c;
    }

    // scale, mask -> NEG
#pragma unroll
    for (int ns = 0; ns < 4; ++ns) {
#pragma unroll
      for (int j = 0; j < 8; ++j) {
        int qr = q0 + j + 8 * hi;
        int key = kc + ns * 16 + nl;
        int mv = mrow[(size_t)qr * Ss + key];
        float sv = sc[ns][j] * scale;
        sc[ns][j] = (mv == 1) ? sv : NEG;
      }
    }

    // online softmax: row max (shfl across 16-lane half) + history
#pragma unroll
    for (int j = 0; j < 8; ++j) {
      float v = fmaxf(fmaxf(sc[0][j], sc[1][j]), fmaxf(sc[2][j], sc[3][j]));
      v = fmaxf(v, __shfl_xor(v, 1, 16));
      v = fmaxf(v, __shfl_xor(v, 2, 16));
      v = fmaxf(v, __shfl_xor(v, 4, 16));
      v = fmaxf(v, __shfl_xor(v, 8, 16));
      float mn = fmaxf(m[j], v);
      float corr = __expf(m[j] - mn);
      m[j] = mn;
      l[j] *= corr;
#pragma unroll
      for (int du = 0; du < 4; ++du) o[du][j] *= corr;
    }

    // P = exp(s-m): row sums + bf16 tile to LDS (C-layout -> A-layout)
    float rsum[8];
#pragma unroll
    for (int j = 0; j < 8; ++j) rsum[j] = 0.0f;
#pragma unroll
    for (int ns = 0; ns < 4; ++ns) {
#pragma unroll
      for (int j = 0; j < 8; ++j) {
        float p = __expf(sc[ns][j] - m[j]);
        rsum[j] += p;
        p_lds[w][(j + 8 * hi) * 64 + ns * 16 + nl] = f2bf(p);
      }
    }
#pragma unroll
    for (int j = 0; j < 8; ++j) {
      float v = rsum[j];
      v += __shfl_xor(v, 1, 16);
      v += __shfl_xor(v, 2, 16);
      v += __shfl_xor(v, 4, 16);
      v += __shfl_xor(v, 8, 16);
      l[j] += v;
    }

    // reload P as A-frags
    v16bf ap0, ap1;
    {
      int r = lane & 15, g8 = (lane >> 4) * 8;
      const ushort_t* base = &p_lds[w][r * 64];
      Frag16 f;
      f.p.lo = *(const v4u*)(base + g8);
      f.p.hi = *(const v4u*)(base + 16 + g8);
      ap0 = f.v;
      f.p.lo = *(const v4u*)(base + 32 + g8);
      f.p.hi = *(const v4u*)(base + 48 + g8);
      ap1 = f.v;
    }

    // O += P(16x64) @ V(64x64), B-operand from vt (contiguous)
#pragma unroll
    for (int du = 0; du < 4; ++du) {
      int d0 = du * 16;
      v16bf bv0 = load_b_frag(vtp, Ss, kc, d0, lane);
      v16bf bv1 = load_b_frag(vtp, Ss, kc + 32, d0, lane);
      o[du] = wmma_bf16(ap0, bv0, o[du]);
      o[du] = wmma_bf16(ap1, bv1, o[du]);
    }
  }

  // normalize + store with the reference reshape quirk
#pragma unroll
  for (int du = 0; du < 4; ++du) {
#pragma unroll
    for (int j = 0; j < 8; ++j) {
      int qr = q0 + j + 8 * hi;
      int d = du * 16 + nl;
      float v = o[du][j] / l[j];
      size_t idx = (size_t)b * Ss * Dm + (size_t)h * Ss * HD +
                   (size_t)qr * HD + d;
      ctx[idx] = f2bf(v);
    }
  }
}

// ---------------------------------------------------------------------------
// Host-side orchestration
// ---------------------------------------------------------------------------
extern "C" void kernel_launch(void* const* d_in, const int* in_sizes, int n_in,
                              void* d_out, int out_size, void* d_ws, size_t ws_size,
                              hipStream_t stream) {
  const float* x    = (const float*)d_in[0];
  const int*   mask = (const int*)d_in[1];
  const float* W0   = (const float*)d_in[2];
  const float* b0   = (const float*)d_in[3];
  const float* W1   = (const float*)d_in[4];
  const float* b1   = (const float*)d_in[5];
  const float* W2   = (const float*)d_in[6];
  const float* b2   = (const float*)d_in[7];
  const float* Wo   = (const float*)d_in[8];
  const float* bo   = (const float*)d_in[9];
  float* out = (float*)d_out;

  ushort_t* ws = (ushort_t*)d_ws;
  size_t off = 0;
  ushort_t* xb  = ws + off; off += (size_t)Mrows * Dm;
  ushort_t* w0t = ws + off; off += (size_t)Dm * Dm;
  ushort_t* w1t = ws + off; off += (size_t)Dm * Dm;
  ushort_t* w2t = ws + off; off += (size_t)Dm * Dm;
  ushort_t* wot = ws + off; off += (size_t)Dm * Dm;
  ushort_t* qb  = ws + off; off += (size_t)Mrows * Dm;
  ushort_t* kkb = ws + off; off += (size_t)Mrows * Dm;
  ushort_t* vtb = ws + off; off += (size_t)Mrows * Dm;  // [B,H,64,S]
  ushort_t* ctx = ws + off; off += (size_t)Mrows * Dm;

  convert_f32_bf16<<<2048, 256, 0, stream>>>(x, xb, Mrows * Dm);
  dim3 tg(Dm / 32, Dm / 32);
  transpose_convert<<<tg, 256, 0, stream>>>(W0, w0t);
  transpose_convert<<<tg, 256, 0, stream>>>(W1, w1t);
  transpose_convert<<<tg, 256, 0, stream>>>(W2, w2t);
  transpose_convert<<<tg, 256, 0, stream>>>(Wo, wot);

  dim3 gg(Dm / 128, Mrows / 128);  // (8, 32)
  gemm_bias_relu<0><<<gg, 256, 0, stream>>>(xb, w0t, b0, (void*)qb,  Mrows, Dm, Dm);
  gemm_bias_relu<0><<<gg, 256, 0, stream>>>(qb, w1t, b1, (void*)kkb, Mrows, Dm, Dm);
  gemm_bias_relu<1><<<gg, 256, 0, stream>>>(qb, w2t, b2, (void*)vtb, Mrows, Dm, Dm);

  dim3 ga(Ss / 64, Hh, Bb);  // (32, 16, 2)
  attention_kernel<<<ga, 128, 0, stream>>>(qb, kkb, vtb, mask, ctx);

  gemm_bias_relu<2><<<gg, 256, 0, stream>>>(ctx, wot, bo, (void*)out, Mrows, Dm, Dm);
}
